// SwinTransformerBlock_31756988187267
// MI455X (gfx1250) — matrix operand order repaired and checked by reading, not compiled, not run on hardware
//
#include <hip/hip_runtime.h>
#include <hip/hip_bf16.h>

typedef _Float16 f16;
typedef __attribute__((ext_vector_type(16))) _Float16 v16h;
typedef __attribute__((ext_vector_type(8)))  _Float16 v8h;
typedef __attribute__((ext_vector_type(4)))  _Float16 v4h;
typedef __attribute__((ext_vector_type(8)))  float    v8f;

#define DIMC 128
#define NHEADS 4
#define NWIN 49
#define SHIFT3 3
#define HW 56
#define NTOK 100352   /* 32*56*56 */
#define MLPH 512
#define LNEPS 1e-5f

#if __has_builtin(__builtin_amdgcn_global_load_async_to_lds_b128)
#define HAVE_ASYNC_LDS 1
#else
#define HAVE_ASYNC_LDS 0
#endif

#if HAVE_ASYNC_LDS
typedef int i32x4 __attribute__((vector_size(16)));
typedef __attribute__((address_space(1))) i32x4 as1_i32x4;
typedef __attribute__((address_space(3))) i32x4 as3_i32x4;

// async 16-byte global -> LDS copy (tracked by ASYNCcnt)
__device__ __forceinline__ void async_copy16(const f16* gsrc, f16* ldst) {
  __builtin_amdgcn_global_load_async_to_lds_b128(
      (as1_i32x4*)(void*)(gsrc), (as3_i32x4*)(void*)(ldst), 0, 0);
}
#endif

__device__ __forceinline__ void wait_async0() {
#if __has_builtin(__builtin_amdgcn_s_wait_asynccnt)
  __builtin_amdgcn_s_wait_asynccnt(0);
#else
  asm volatile("s_wait_asynccnt 0x0" ::: "memory");
#endif
}

union Frag {
  v16h v;
  v8h  h2[2];
  f16  e[16];
};

__device__ __forceinline__ v8f wmma16(const Frag& a, const Frag& b, v8f c) {
  return __builtin_amdgcn_wmma_f32_16x16x32_f16(false, a.v, false, b.v,
                                                (short)0, c, false, false);
}

__device__ __forceinline__ float wave_sum(float v) {
#pragma unroll
  for (int m = 16; m >= 1; m >>= 1) v += __shfl_xor(v, m, 32);
  return v;
}
__device__ __forceinline__ float half_max(float v) {
#pragma unroll
  for (int m = 8; m >= 1; m >>= 1) v = fmaxf(v, __shfl_xor(v, m, 32));
  return v;
}
__device__ __forceinline__ float half_sum(float v) {
#pragma unroll
  for (int m = 8; m >= 1; m >>= 1) v += __shfl_xor(v, m, 32);
  return v;
}

// windowed-row index -> flat token index (same map for gather and scatter)
__device__ __forceinline__ int win_row_to_token(int r) {
  const int win = r / NWIN;
  const int t = r - win * NWIN;
  const int b = win >> 6;  // 8x8 windows per image
  const int w2 = win & 63;
  const int wr = w2 >> 3, wc = w2 & 7;
  const int ii = t / 7, jj = t - ii * 7;
  int sr = wr * 7 + ii + SHIFT3; if (sr >= HW) sr -= HW;
  int sc = wc * 7 + jj + SHIFT3; if (sc >= HW) sc -= HW;
  return b * (HW * HW) + sr * HW + sc;
}

// ---------------------------------------------------------------------------
// LayerNorm (one wave per 128-wide row), optional shifted-window gather.
// ---------------------------------------------------------------------------
__global__ __launch_bounds__(256) void ln_kernel(const float* __restrict__ src,
                                                 const float* __restrict__ gw,
                                                 const float* __restrict__ bw,
                                                 f16* __restrict__ dst,
                                                 int gather) {
  const int lane = threadIdx.x & 31;
  const int r = blockIdx.x * 8 + (threadIdx.x >> 5);
  const int srow = gather ? win_row_to_token(r) : r;
  const float4 xv = *(const float4*)(src + (size_t)srow * DIMC + lane * 4);
  float s = xv.x + xv.y + xv.z + xv.w;
  s = wave_sum(s);
  const float mu = s * (1.0f / DIMC);
  const float dx = xv.x - mu, dy = xv.y - mu, dz = xv.z - mu, dw = xv.w - mu;
  float q = dx * dx + dy * dy + dz * dz + dw * dw;
  q = wave_sum(q);
  const float rs = rsqrtf(q * (1.0f / DIMC) + LNEPS);
  const float4 gv = *(const float4*)(gw + lane * 4);
  const float4 bv = *(const float4*)(bw + lane * 4);
  v4h o;
  o[0] = (f16)(dx * rs * gv.x + bv.x);
  o[1] = (f16)(dy * rs * gv.y + bv.y);
  o[2] = (f16)(dz * rs * gv.z + bv.z);
  o[3] = (f16)(dw * rs * gv.w + bv.w);
  *(v4h*)(dst + (size_t)r * DIMC + lane * 4) = o;
}

// ---------------------------------------------------------------------------
// Generic GEMM: out(M x ND) = A(M x KD, f16) @ W(ND x KD, f32)^T + bias
// 128x128 tile / 256 threads (8 waves of 32x64), K-step 32, double-buffered
// LDS with async global->LDS staging for the A tile.
// EPI: 0 = qkv(f16, ld 384), 1 = proj(window scatter + residual, f32),
//      2 = fc1(exact GELU, f16), 3 = fc2(residual, f32)
// ---------------------------------------------------------------------------
template <int KD, int ND, int EPI>
__global__ __launch_bounds__(256) void gemm_wmma_kernel(
    const f16* __restrict__ A, const float* __restrict__ W,
    const float* __restrict__ bias, const float* __restrict__ resid,
    float* __restrict__ outf, f16* __restrict__ outh) {
  __shared__ __align__(32) f16 As[2][128 * 32];
  __shared__ __align__(32) f16 Bs[2][128 * 32];  // [n][k]
  const int tid = threadIdx.x;
  const int lane = tid & 31;
  const int w = tid >> 5;
  const int wm = w >> 1, wn = w & 1;
  const int g = lane >> 4, ln = lane & 15;
  const int m0 = blockIdx.x * 128;
  const int n0 = blockIdx.y * 128;

  v8f acc[2][4];
#pragma unroll
  for (int s = 0; s < 2; ++s)
#pragma unroll
    for (int j = 0; j < 4; ++j)
#pragma unroll
      for (int e = 0; e < 8; ++e) acc[s][j][e] = 0.f;

  const int srow = tid >> 1;       // 0..127
  const int scol = (tid & 1) * 16; // 0 or 16
  const f16* abase = A + (size_t)(m0 + srow) * KD + scol;
  const float* wbase = W + (size_t)(n0 + srow) * KD + scol;
  const int soff = srow * 32 + scol;
  const int nsteps = KD / 32;

  // ---- stage step 0 ----
  {
#if HAVE_ASYNC_LDS
    async_copy16(abase, &As[0][soff]);
    async_copy16(abase + 8, &As[0][soff + 8]);
#else
    *(v8h*)(&As[0][soff]) = *(const v8h*)abase;
    *(v8h*)(&As[0][soff + 8]) = *(const v8h*)(abase + 8);
#endif
    const float4 w0 = *(const float4*)(wbase + 0);
    const float4 w1 = *(const float4*)(wbase + 4);
    const float4 w2 = *(const float4*)(wbase + 8);
    const float4 w3 = *(const float4*)(wbase + 12);
    f16* bd = &Bs[0][soff];
    bd[0] = (f16)w0.x; bd[1] = (f16)w0.y; bd[2] = (f16)w0.z; bd[3] = (f16)w0.w;
    bd[4] = (f16)w1.x; bd[5] = (f16)w1.y; bd[6] = (f16)w1.z; bd[7] = (f16)w1.w;
    bd[8] = (f16)w2.x; bd[9] = (f16)w2.y; bd[10] = (f16)w2.z; bd[11] = (f16)w2.w;
    bd[12] = (f16)w3.x; bd[13] = (f16)w3.y; bd[14] = (f16)w3.z; bd[15] = (f16)w3.w;
#if HAVE_ASYNC_LDS
    wait_async0();
#endif
    __syncthreads();
  }

  int p = 0;
  for (int step = 0; step < nsteps; ++step) {
    const bool hasNext = (step + 1) < nsteps;
    v8h na0, na1;
    float4 nw0, nw1, nw2, nw3;
    if (hasNext) {
      const f16* asrc = abase + (step + 1) * 32;
#if HAVE_ASYNC_LDS
      async_copy16(asrc, &As[p ^ 1][soff]);
      async_copy16(asrc + 8, &As[p ^ 1][soff + 8]);
#else
      na0 = *(const v8h*)asrc;
      na1 = *(const v8h*)(asrc + 8);
#endif
      const float* wsrc = wbase + (step + 1) * 32;
      nw0 = *(const float4*)(wsrc + 0);
      nw1 = *(const float4*)(wsrc + 4);
      nw2 = *(const float4*)(wsrc + 8);
      nw3 = *(const float4*)(wsrc + 12);
    }

    Frag af[2], bf[4];
#pragma unroll
    for (int s = 0; s < 2; ++s) {
      const f16* ap = &As[p][(wm * 32 + s * 16 + ln) * 32];
      af[s].h2[0] = *(const v8h*)(ap + g * 8);
      af[s].h2[1] = *(const v8h*)(ap + 16 + g * 8);
    }
#pragma unroll
    for (int j = 0; j < 4; ++j) {
      const f16* bp = &Bs[p][(wn * 64 + j * 16 + ln) * 32];
      bf[j].h2[0] = *(const v8h*)(bp + g * 8);
      bf[j].h2[1] = *(const v8h*)(bp + 16 + g * 8);
    }
#pragma unroll
    for (int s = 0; s < 2; ++s)
#pragma unroll
      for (int j = 0; j < 4; ++j)
        acc[s][j] = wmma16(af[s], bf[j], acc[s][j]);

    if (hasNext) {
#if !HAVE_ASYNC_LDS
      *(v8h*)(&As[p ^ 1][soff]) = na0;
      *(v8h*)(&As[p ^ 1][soff + 8]) = na1;
#endif
      f16* bd = &Bs[p ^ 1][soff];
      bd[0] = (f16)nw0.x; bd[1] = (f16)nw0.y; bd[2] = (f16)nw0.z; bd[3] = (f16)nw0.w;
      bd[4] = (f16)nw1.x; bd[5] = (f16)nw1.y; bd[6] = (f16)nw1.z; bd[7] = (f16)nw1.w;
      bd[8] = (f16)nw2.x; bd[9] = (f16)nw2.y; bd[10] = (f16)nw2.z; bd[11] = (f16)nw2.w;
      bd[12] = (f16)nw3.x; bd[13] = (f16)nw3.y; bd[14] = (f16)nw3.z; bd[15] = (f16)nw3.w;
#if HAVE_ASYNC_LDS
      wait_async0();
#endif
      __syncthreads();
      p ^= 1;
    }
  }

  // epilogue. C layout: row = vgpr + 8*(lane>=16), col = lane%16
#pragma unroll
  for (int s = 0; s < 2; ++s) {
#pragma unroll
    for (int j = 0; j < 4; ++j) {
      const int n = n0 + wn * 64 + j * 16 + ln;
      const float bn = bias[n];
#pragma unroll
      for (int v = 0; v < 8; ++v) {
        const int m = m0 + wm * 32 + s * 16 + v + g * 8;
        const float val = acc[s][j][v] + bn;
        if (EPI == 0) {
          outh[(size_t)m * 384 + n] = (f16)val;
        } else if (EPI == 1) {
          const int tok = win_row_to_token(m);
          const size_t idx = (size_t)tok * DIMC + n;
          outf[idx] = resid[idx] + val;
        } else if (EPI == 2) {
          const float ge = 0.5f * val * (1.0f + erff(val * 0.70710678118654752f));
          outh[(size_t)m * MLPH + n] = (f16)ge;
        } else {
          const size_t idx = (size_t)m * DIMC + n;
          outf[idx] = resid[idx] + val;
        }
      }
    }
  }
}

// ---------------------------------------------------------------------------
// Windowed attention: 1 block per window (128 threads), 1 wave per head.
// rpb table staged in LDS; bias add fully branchless (invalid rows/cols are
// either masked afterwards or never stored); matmuls via WMMA f16.
// ---------------------------------------------------------------------------
__global__ __launch_bounds__(128) void attn_kernel(const f16* __restrict__ qkv,
                                                   const float* __restrict__ rpb,
                                                   f16* __restrict__ out) {
  __shared__ __align__(32) f16 P[NHEADS][64 * 64];   // probabilities
  __shared__ __align__(32) f16 Vt[NHEADS][32 * 64];  // V^T [dim][tok]
  __shared__ float rpbs[169 * NHEADS];               // 2.7KB bias table
  const int lane = threadIdx.x & 31;
  const int h = threadIdx.x >> 5;
  const int g = lane >> 4, ln = lane & 15;
  const long rowbase = (long)blockIdx.x * NWIN;
  const f16* wq = qkv + rowbase * 384;
  const float scale = 0.17677669529663687f;  // 1/sqrt(32)

  for (int i = threadIdx.x; i < 169 * NHEADS; i += 128) rpbs[i] = rpb[i];

  // stage V transposed (pad tokens 49..63 with zero)
  for (int t = lane; t < 64; t += 32) {
    if (t < NWIN) {
      const f16* vp = wq + t * 384 + 256 + h * 32;
#pragma unroll
      for (int d = 0; d < 32; ++d) Vt[h][d * 64 + t] = vp[d];
    } else {
#pragma unroll
      for (int d = 0; d < 32; ++d) Vt[h][d * 64 + t] = (f16)0.f;
    }
  }
  __syncthreads();

  // K as B-fragments (B[k][n] = K[token n][dim k]; dims contiguous per token)
  Frag bk[4];
  int tnj[4], inj[4], jnj[4];
#pragma unroll
  for (int j = 0; j < 4; ++j) {
    tnj[j] = j * 16 + ln;
    inj[j] = tnj[j] / 7;
    jnj[j] = tnj[j] - inj[j] * 7;
#pragma unroll
    for (int e = 0; e < 16; ++e) bk[j].e[e] = (f16)0.f;
    if (tnj[j] < NWIN) {
      const f16* kp = wq + tnj[j] * 384 + 128 + h * 32 + g * 8;
      bk[j].h2[0] = *(const v8h*)kp;
      bk[j].h2[1] = *(const v8h*)(kp + 16);
    }
  }

  // scores + masked softmax, one 16-row tile at a time
  for (int i = 0; i < 4; ++i) {
    Frag aq;
#pragma unroll
    for (int e = 0; e < 16; ++e) aq.e[e] = (f16)0.f;
    const int tml = i * 16 + ln;
    if (tml < NWIN) {
      const f16* qp = wq + tml * 384 + h * 32 + g * 8;
      aq.h2[0] = *(const v8h*)qp;
      aq.h2[1] = *(const v8h*)(qp + 16);
    }
    v8f sc[4];
#pragma unroll
    for (int j = 0; j < 4; ++j) {
      v8f z;
#pragma unroll
      for (int e = 0; e < 8; ++e) z[e] = 0.f;
      sc[j] = wmma16(aq, bk[j], z);
    }
#pragma unroll
    for (int v = 0; v < 8; ++v) {
      const int tm = i * 16 + v + g * 8;
      const int im = tm / 7, jm = tm - im * 7;
      float vals[4];
      float mx = -3.0e38f;
#pragma unroll
      for (int j = 0; j < 4; ++j) {
        // unconditional clamped LDS bias gather: rows tm>=49 are never stored
        // and cols tn>=49 get masked below, so a garbage bias is harmless.
        int idx = (im - inj[j] + 6) * 13 + (jm - jnj[j] + 6);
        idx = min(max(idx, 0), 168);
        const float bb = rpbs[idx * NHEADS + h];
        float xx = sc[j][v] * scale + bb;
        xx = (tnj[j] < NWIN) ? xx : -3.0e38f;
        vals[j] = xx;
        mx = fmaxf(mx, xx);
      }
      mx = half_max(mx);
      float ssum = 0.f;
#pragma unroll
      for (int j = 0; j < 4; ++j) {
        const float e = (vals[j] <= -1.0e37f) ? 0.f : __expf(vals[j] - mx);
        vals[j] = e;
        ssum += e;
      }
      ssum = half_sum(ssum);
      const float inv = 1.0f / ssum;
#pragma unroll
      for (int j = 0; j < 4; ++j)
        P[h][tm * 64 + tnj[j]] = (f16)(vals[j] * inv);
    }
  }
  __syncthreads();

  // out = P(64x64) @ V(64x32): two k-steps of 32
  Frag bv[2][2];
#pragma unroll
  for (int ks = 0; ks < 2; ++ks)
#pragma unroll
    for (int jt = 0; jt < 2; ++jt) {
      const f16* vp = &Vt[h][(jt * 16 + ln) * 64 + ks * 32 + g * 8];
      bv[ks][jt].h2[0] = *(const v8h*)vp;
      bv[ks][jt].h2[1] = *(const v8h*)(vp + 16);
    }
  for (int i = 0; i < 4; ++i) {
    Frag ap[2];
#pragma unroll
    for (int ks = 0; ks < 2; ++ks) {
      const f16* pp = &P[h][(i * 16 + ln) * 64 + ks * 32 + g * 8];
      ap[ks].h2[0] = *(const v8h*)pp;
      ap[ks].h2[1] = *(const v8h*)(pp + 16);
    }
    v8f acc[2];
#pragma unroll
    for (int jt = 0; jt < 2; ++jt) {
#pragma unroll
      for (int e = 0; e < 8; ++e) acc[jt][e] = 0.f;
      acc[jt] = wmma16(ap[0], bv[0][jt], acc[jt]);
      acc[jt] = wmma16(ap[1], bv[1][jt], acc[jt]);
    }
#pragma unroll
    for (int jt = 0; jt < 2; ++jt)
#pragma unroll
      for (int v = 0; v < 8; ++v) {
        const int tm = i * 16 + v + g * 8;
        if (tm < NWIN)
          out[(rowbase + tm) * DIMC + h * 32 + jt * 16 + ln] = (f16)acc[jt][v];
      }
  }
}

// ---------------------------------------------------------------------------
extern "C" void kernel_launch(void* const* d_in, const int* in_sizes, int n_in,
                              void* d_out, int out_size, void* d_ws,
                              size_t ws_size, hipStream_t stream) {
  const float* x = (const float*)d_in[0];
  const float* n1g = (const float*)d_in[1];
  const float* n1b = (const float*)d_in[2];
  const float* qkv_w = (const float*)d_in[3];
  const float* qkv_b = (const float*)d_in[4];
  const float* rpb = (const float*)d_in[5];
  const float* proj_w = (const float*)d_in[6];
  const float* proj_b = (const float*)d_in[7];
  const float* n2g = (const float*)d_in[8];
  const float* n2b = (const float*)d_in[9];
  const float* fc1_w = (const float*)d_in[10];
  const float* fc1_b = (const float*)d_in[11];
  const float* fc2_w = (const float*)d_in[12];
  const float* fc2_b = (const float*)d_in[13];
  float* out = (float*)d_out;

  // workspace (aliased along dependency chain):
  //  buf0 (NTOK*128 f16): xw -> attn -> ln2
  //  buf1 (NTOK*512 f16): qkv (NTOK*384, ld 384) -> hid
  //  buf2 (NTOK*128 f32): x_new
  char* wsb = (char*)d_ws;
  f16* xw = (f16*)wsb;
  f16* qkvb = (f16*)(wsb + (size_t)NTOK * DIMC * 2);
  float* xnew = (float*)(wsb + (size_t)NTOK * DIMC * 2 + (size_t)NTOK * MLPH * 2);
  f16* attn = xw;
  f16* ln2 = xw;
  f16* hid = qkvb;

  ln_kernel<<<NTOK / 8, 256, 0, stream>>>(x, n1g, n1b, xw, 1);
  gemm_wmma_kernel<128, 384, 0><<<dim3(NTOK / 128, 3), 256, 0, stream>>>(
      xw, qkv_w, qkv_b, nullptr, nullptr, qkvb);
  attn_kernel<<<2048, 128, 0, stream>>>(qkvb, rpb, attn);
  gemm_wmma_kernel<128, 128, 1><<<dim3(NTOK / 128, 1), 256, 0, stream>>>(
      attn, proj_w, proj_b, x, xnew, nullptr);
  ln_kernel<<<NTOK / 8, 256, 0, stream>>>(xnew, n2g, n2b, ln2, 0);
  gemm_wmma_kernel<128, 512, 2><<<dim3(NTOK / 128, 4), 256, 0, stream>>>(
      ln2, fc1_w, fc1_b, nullptr, nullptr, hid);
  gemm_wmma_kernel<512, 128, 3><<<dim3(NTOK / 128, 1), 256, 0, stream>>>(
      hid, fc2_w, fc2_b, xnew, out, nullptr);
}
